// PreGatingContextualAttentionGate_49667001811646
// MI455X (gfx1250) — matrix operand order repaired and checked by reading, not compiled, not run on hardware
//
#include <hip/hip_runtime.h>
#include <hip/hip_bf16.h>

// ---------------------------------------------------------------------------
// MI455X (gfx1250) implementation: bf16 WMMA (v_wmma_f32_16x16x32_bf16),
// wave32, 256 KB LDS score strips, f32 accumulation everywhere.
// ---------------------------------------------------------------------------

typedef __attribute__((ext_vector_type(16))) __bf16 v16bf;
typedef __attribute__((ext_vector_type(8)))  __bf16 v8bf;
typedef __attribute__((ext_vector_type(8)))  float  v8f;

#define BATCH 4
#define NSEQ  4096
#define DKDIM 256
#define OUTD  128
#define BN    (BATCH * NSEQ)   // 16384 rows
#define SCLD  4100             // LDS row stride (pad: 4*r mod 64 distinct banks)

// Native conversion: backend emits the hardware f32->bf16 cvt (RTNE).
__device__ __forceinline__ __bf16 f2bf(float f) { return (__bf16)f; }

// D = A*B + C, 16x16x32 bf16 -> f32  (ISA VOP3P op 98)
__device__ __forceinline__ v8f wmma_bf16(v16bf a, v16bf b, v8f c) {
  return __builtin_amdgcn_wmma_f32_16x16x32_bf16(false, a, false, b,
                                                 (short)0, c, false, false);
}

// A-operand (16x32 bf16, contiguous k order, same as load_b_bf16): lane m=lane&15 -> row m;
// lanes 0-15 hold K=k0..k0+15, lanes 16-31 hold K=k0+16..k0+31.
__device__ __forceinline__ v16bf load_a_bf16(const __bf16* src, int ld, int k0, int lane) {
  int half = lane >> 4, m = lane & 15;
  const __bf16* p = src + (size_t)m * ld + k0 + half * 16;
  v8bf lo = *reinterpret_cast<const v8bf*>(p);
  v8bf hi = *reinterpret_cast<const v8bf*>(p + 8);
  return __builtin_shufflevector(lo, hi, 0, 1, 2, 3, 4, 5, 6, 7,
                                 8, 9, 10, 11, 12, 13, 14, 15);
}

// B-operand (32x16 bf16): lane n=lane&15 -> column n; lanes 0-15 hold K=0..15,
// lanes 16-31 hold K=16..31. src is stored transposed: src[n][k] contiguous in k.
__device__ __forceinline__ v16bf load_b_bf16(const __bf16* src, int ld, int k0, int lane) {
  int half = lane >> 4, n = lane & 15;
  return *reinterpret_cast<const v16bf*>(src + (size_t)n * ld + k0 + half * 16);
}

// ---------------------------------------------------------------------------
// Kernel 0: transpose f32 weight [fin][fout] -> bf16 [fout][fin]
// ---------------------------------------------------------------------------
__global__ void wt_transpose_kernel(const float* __restrict__ W,
                                    __bf16* __restrict__ Wt, int fin, int fout) {
  int idx = blockIdx.x * blockDim.x + threadIdx.x;
  if (idx >= fin * fout) return;
  int o = idx % fout;   // consecutive threads read consecutive W columns (coalesced)
  int i = idx / fout;
  Wt[(size_t)o * fin + i] = f2bf(W[(size_t)i * fout + o]);
}

// ---------------------------------------------------------------------------
// Kernel 1: Y = X @ W + b for the 256->256 projections, with fused epilogues.
// Workgroup: 8 waves; rows [bid*32, bid*32+32), all 256 output columns.
// X tile staged once in LDS as bf16 (avoids 4x duplicated conversions).
// ---------------------------------------------------------------------------
__global__ __launch_bounds__(256)
void qkv_gemm_kernel(const float*  __restrict__ X,    // [BN][256] f32
                     const __bf16* __restrict__ Wt,   // [256][256] bf16 (transposed)
                     const float*  __restrict__ bias, // [256]
                     float*  __restrict__ out_f32,    // nullable [BN][256]
                     __bf16* __restrict__ out_bf,     // nullable [BN][256]
                     __bf16* __restrict__ out_tanh,   // nullable [BN][256]
                     __bf16* __restrict__ out_T)      // nullable [B][256][NSEQ]
{
  __shared__ __bf16 Xs[32 * DKDIM];                   // 16 KB
  const int lane = threadIdx.x & 31;
  const int wave = threadIdx.x >> 5;
  const int half = lane >> 4;
  const int np   = lane & 15;
  const size_t rowbase = (size_t)blockIdx.x * 32;
  const int mbase = (int)rowbase + (wave >> 2) * 16;
  const int cbase = (wave & 3) * 64;

  for (int i = threadIdx.x; i < 32 * DKDIM; i += 256)
    Xs[i] = f2bf(X[rowbase * DKDIM + i]);
  __syncthreads();
  const __bf16* xa = Xs + (size_t)(wave >> 2) * 16 * DKDIM;

  v8f acc[4] = {};
#pragma unroll
  for (int kc = 0; kc < 8; ++kc) {
    int k0 = kc * 32;
    v16bf a = load_a_bf16(xa, DKDIM, k0, lane);
#pragma unroll
    for (int t = 0; t < 4; ++t) {
      v16bf b = load_b_bf16(Wt + (size_t)(cbase + t * 16) * DKDIM, DKDIM, k0, lane);
      acc[t] = wmma_bf16(a, b, acc[t]);
    }
  }
#pragma unroll
  for (int t = 0; t < 4; ++t) {
    int col = cbase + t * 16 + np;
    float bv = bias[col];
#pragma unroll
    for (int i = 0; i < 8; ++i) {
      int row = mbase + i + half * 8;            // C layout: M = vgpr + 8*(lane>=16)
      float v = acc[t][i] + bv;
      if (out_f32)  out_f32[(size_t)row * DKDIM + col] = v;
      if (out_bf)   out_bf [(size_t)row * DKDIM + col] = f2bf(v);
      if (out_tanh) out_tanh[(size_t)row * DKDIM + col] = f2bf(tanhf(v));
      if (out_T) {                                // V stored transposed per batch
        int b_ = row >> 12, key = row & (NSEQ - 1);
        out_T[((size_t)b_ * DKDIM + col) * NSEQ + key] = f2bf(v);
      }
    }
  }
}

// ---------------------------------------------------------------------------
// Kernel 2: scores -> softmax -> attn (output) -> Q_hat = attn @ V.
// One workgroup = 16 query rows; 16x4096 f32 score strip lives in LDS (256 KB).
// The softmax normalize pass rewrites each LDS row in place as bf16 so the
// Q_hat WMMA loop does pure ds_load -> v_wmma with no conversions.
// K/tanhK/V are 6 MB bf16 per batch -> resident in the 192 MB L2.
// ---------------------------------------------------------------------------
__global__ __launch_bounds__(256)
void attention_kernel(const __bf16* __restrict__ Qb,  // [BN][256]
                      const __bf16* __restrict__ Tq,  // tanh(Q) [BN][256]
                      const __bf16* __restrict__ Kb,  // [BN][256]
                      const __bf16* __restrict__ Tk,  // tanh(K) [BN][256]
                      const __bf16* __restrict__ Vt,  // [B][256][NSEQ]
                      float* __restrict__ attn_out,   // [B][NSEQ][NSEQ]
                      float* __restrict__ qhat)       // [BN][256]
{
  extern __shared__ float sc[];                  // [16][SCLD]
  const int lane = threadIdx.x & 31;
  const int wave = threadIdx.x >> 5;
  const int half = lane >> 4;
  const int np   = lane & 15;
  const int b    = blockIdx.y;
  const int qbase = blockIdx.x * 16;
  const size_t qrow0 = (size_t)b * NSEQ + qbase;

  // Keep both A operands (Q tile, tanhQ tile, 16x256 each) resident: 128 VGPRs.
  v16bf aQ[8], aT[8];
#pragma unroll
  for (int kc = 0; kc < 8; ++kc) {
    aQ[kc] = load_a_bf16(Qb + qrow0 * DKDIM, DKDIM, kc * 32, lane);
    aT[kc] = load_a_bf16(Tq + qrow0 * DKDIM, DKDIM, kc * 32, lane);
  }
  const __bf16* Kbase  = Kb + (size_t)b * NSEQ * DKDIM;
  const __bf16* Tkbase = Tk + (size_t)b * NSEQ * DKDIM;

  // ---- scores: each wave handles 32 interleaved 16-key tiles -------------
  for (int nt = wave; nt < NSEQ / 16; nt += 8) {
    v8f accS = {}, accP = {};
    const __bf16* kcol = Kbase  + (size_t)(nt * 16) * DKDIM;
    const __bf16* tcol = Tkbase + (size_t)(nt * 16) * DKDIM;
#pragma unroll
    for (int kc = 0; kc < 8; ++kc) {
      int k0 = kc * 32;
      accS = wmma_bf16(aQ[kc], load_b_bf16(kcol, DKDIM, k0, lane), accS);
      accP = wmma_bf16(aT[kc], load_b_bf16(tcol, DKDIM, k0, lane), accP);
    }
#pragma unroll
    for (int i = 0; i < 8; ++i) {
      int r = i + half * 8;
      // QK * (tanhQK+1)*0.5 / sqrt(256)
      sc[r * SCLD + nt * 16 + np] = accS[i] * (accP[i] + 1.0f) * 0.03125f;
    }
  }
  __syncthreads();

  // ---- softmax: wave w owns rows {2w, 2w+1}; wave32 shuffle reductions ---
#pragma unroll
  for (int rr = 0; rr < 2; ++rr) {
    int r = wave * 2 + rr;
    float mx = -3.0e38f;
    for (int i = lane; i < NSEQ; i += 32) mx = fmaxf(mx, sc[r * SCLD + i]);
#pragma unroll
    for (int o = 16; o > 0; o >>= 1) mx = fmaxf(mx, __shfl_xor(mx, o, 32));
    float sum = 0.0f;
    for (int i = lane; i < NSEQ; i += 32) {
      float p = __expf(sc[r * SCLD + i] - mx);
      sc[r * SCLD + i] = p;
      sum += p;
    }
#pragma unroll
    for (int o = 16; o > 0; o >>= 1) sum += __shfl_xor(sum, o, 32);
    float inv = 1.0f / sum;
    float* arow = attn_out + (qrow0 + r) * NSEQ;
    __bf16* brow = reinterpret_cast<__bf16*>(&sc[(size_t)r * SCLD]);
    // In-place f32 -> bf16 rewrite: within each strided iteration the 32 f32
    // reads (slots 32j..32j+31) precede the bf16 stores (touch slots
    // 16j..16j+15, all already consumed). Row is private to this wave.
    for (int i = lane; i < NSEQ; i += 32) {
      float v = sc[r * SCLD + i] * inv;
      arow[i] = v;                                // coalesced attn store
      brow[i] = f2bf(v);
    }
  }
  __syncthreads();

  // ---- Q_hat = attn(16x4096) @ V(4096x256): wave w owns cols 32w..32w+31 --
  const __bf16* vtb = Vt + (size_t)b * DKDIM * NSEQ;
  const __bf16* abase = reinterpret_cast<const __bf16*>(sc);  // bf16 rows, stride 2*SCLD
  const int c0 = wave * 32;
  v8f acc0 = {}, acc1 = {};
  for (int kc = 0; kc < NSEQ / 32; ++kc) {
    int k0 = kc * 32;
    v16bf a = load_a_bf16(abase, SCLD * 2, k0, lane);
    acc0 = wmma_bf16(a, load_b_bf16(vtb + (size_t)c0 * NSEQ,        NSEQ, k0, lane), acc0);
    acc1 = wmma_bf16(a, load_b_bf16(vtb + (size_t)(c0 + 16) * NSEQ, NSEQ, k0, lane), acc1);
  }
#pragma unroll
  for (int i = 0; i < 8; ++i) {
    int r = i + half * 8;
    qhat[(qrow0 + r) * DKDIM + c0 + np]      = acc0[i];
    qhat[(qrow0 + r) * DKDIM + c0 + 16 + np] = acc1[i];
  }
}

// ---------------------------------------------------------------------------
// Kernel 3: contextual attention gate. One workgroup = 16 rows; 8 waves each
// own one 16-column tile of the 128-wide outputs.
// ---------------------------------------------------------------------------
__global__ __launch_bounds__(256)
void cag_kernel(const float* __restrict__ Q,      // [BN][256]
                const float* __restrict__ Qhat,   // [BN][256]
                const __bf16* __restrict__ Wt1, const float* __restrict__ bb1,
                const __bf16* __restrict__ Wt2, const float* __restrict__ bb2,
                const __bf16* __restrict__ Wt3, const float* __restrict__ bb3,
                const __bf16* __restrict__ Wtf, const float* __restrict__ bbf,
                const __bf16* __restrict__ Wtc, const float* __restrict__ bbc,
                const float* __restrict__ g_gamma, const float* __restrict__ g_beta,
                const float* __restrict__ e_gamma, const float* __restrict__ e_beta,
                float* __restrict__ out0)         // [BN][128]
{
  __shared__ __bf16 Qs[16 * DKDIM];
  __shared__ __bf16 Hs[16 * DKDIM];
  __shared__ float  Gpre[16 * OUTD];
  __shared__ float  Epre[16 * OUTD];
  __shared__ __bf16 GEs[16 * OUTD];

  const int lane = threadIdx.x & 31;
  const int wave = threadIdx.x >> 5;
  const int half = lane >> 4;
  const int np   = lane & 15;
  const size_t row0 = (size_t)blockIdx.x * 16;

  for (int i = threadIdx.x; i < 16 * DKDIM; i += 256) {
    Qs[i] = f2bf(Q[row0 * DKDIM + i]);
    Hs[i] = f2bf(Qhat[row0 * DKDIM + i]);
  }
  __syncthreads();

  const int c0 = wave * 16;
  v8f a1 = {}, a2 = {}, a3 = {}, af = {};
#pragma unroll
  for (int kc = 0; kc < 8; ++kc) {
    int k0 = kc * 32;
    v16bf aQ = load_a_bf16(Qs, DKDIM, k0, lane);
    v16bf aH = load_a_bf16(Hs, DKDIM, k0, lane);
    a1 = wmma_bf16(aQ, load_b_bf16(Wt1 + (size_t)c0 * DKDIM, DKDIM, k0, lane), a1);
    a2 = wmma_bf16(aH, load_b_bf16(Wt2 + (size_t)c0 * DKDIM, DKDIM, k0, lane), a2);
    a3 = wmma_bf16(aH, load_b_bf16(Wt3 + (size_t)c0 * DKDIM, DKDIM, k0, lane), a3);
    af = wmma_bf16(aQ, load_b_bf16(Wtf + (size_t)c0 * DKDIM, DKDIM, k0, lane), af);
  }
  float qf[8];
  {
    int col = c0 + np;
    float v1 = bb1[col], v2 = bb2[col], v3 = bb3[col], vf = bbf[col];
#pragma unroll
    for (int i = 0; i < 8; ++i) {
      int r = i + half * 8;
      float g = fmaxf(fmaxf(a1[i] + v1, 0.f) + fmaxf(a2[i] + v2, 0.f), 0.f);
      float e = fmaxf(a3[i] + v3, 0.f);           // relu(relu(x)) == relu(x)
      Gpre[r * OUTD + col] = g;
      Epre[r * OUTD + col] = e;
      qf[i] = fmaxf(af[i] + vf, 0.f);
    }
  }
  __syncthreads();

  // layernorms over the 128-wide rows: wave w owns rows {2w, 2w+1}
#pragma unroll
  for (int rr = 0; rr < 2; ++rr) {
    int r = wave * 2 + rr;
    float gv[4], ev[4], gs = 0.f, es = 0.f;
#pragma unroll
    for (int j = 0; j < 4; ++j) {
      int c = lane + j * 32;
      gv[j] = Gpre[r * OUTD + c]; ev[j] = Epre[r * OUTD + c];
      gs += gv[j]; es += ev[j];
    }
#pragma unroll
    for (int o = 16; o > 0; o >>= 1) { gs += __shfl_xor(gs, o, 32); es += __shfl_xor(es, o, 32); }
    float gmu = gs * (1.f / 128.f), emu = es * (1.f / 128.f);
    float gvr = 0.f, evr = 0.f;
#pragma unroll
    for (int j = 0; j < 4; ++j) {
      float dg = gv[j] - gmu, de = ev[j] - emu;
      gvr += dg * dg; evr += de * de;
    }
#pragma unroll
    for (int o = 16; o > 0; o >>= 1) { gvr += __shfl_xor(gvr, o, 32); evr += __shfl_xor(evr, o, 32); }
    float gr = rsqrtf(gvr * (1.f / 128.f) + 1e-5f);
    float er = rsqrtf(evr * (1.f / 128.f) + 1e-5f);
#pragma unroll
    for (int j = 0; j < 4; ++j) {
      int c = lane + j * 32;
      float G = (gv[j] - gmu) * gr * g_gamma[c] + g_beta[c];
      float E = (ev[j] - emu) * er * e_gamma[c] + e_beta[c];
      GEs[r * OUTD + c] = f2bf(G * E);
    }
  }
  __syncthreads();

  // C = relu((G*E) @ W_c + b_c); out = Qf + C
  v8f ac = {};
#pragma unroll
  for (int kc = 0; kc < 4; ++kc) {
    int k0 = kc * 32;
    v16bf aG = load_a_bf16(GEs, OUTD, k0, lane);
    ac = wmma_bf16(aG, load_b_bf16(Wtc + (size_t)c0 * OUTD, OUTD, k0, lane), ac);
  }
  {
    int col = c0 + np;
    float bv = bbc[col];
#pragma unroll
    for (int i = 0; i < 8; ++i) {
      int r = i + half * 8;
      out0[(row0 + r) * OUTD + col] = qf[i] + fmaxf(ac[i] + bv, 0.f);
    }
  }
}

// ---------------------------------------------------------------------------
extern "C" void kernel_launch(void* const* d_in, const int* in_sizes, int n_in,
                              void* d_out, int out_size, void* d_ws, size_t ws_size,
                              hipStream_t stream) {
  const float* x1   = (const float*)d_in[0];
  const float* x2   = (const float*)d_in[1];
  const float* W_q  = (const float*)d_in[2];
  const float* b_q  = (const float*)d_in[3];
  const float* W_k  = (const float*)d_in[4];
  const float* b_k  = (const float*)d_in[5];
  const float* W_v  = (const float*)d_in[6];
  const float* b_v  = (const float*)d_in[7];
  const float* W1   = (const float*)d_in[8];
  const float* b1   = (const float*)d_in[9];
  const float* W2   = (const float*)d_in[10];
  const float* b2   = (const float*)d_in[11];
  const float* W3   = (const float*)d_in[12];
  const float* b3   = (const float*)d_in[13];
  const float* g_g  = (const float*)d_in[14];
  const float* g_b  = (const float*)d_in[15];
  const float* e_g  = (const float*)d_in[16];
  const float* e_b  = (const float*)d_in[17];
  const float* W_c  = (const float*)d_in[18];
  const float* b_c  = (const float*)d_in[19];
  const float* W_f  = (const float*)d_in[20];
  const float* b_f  = (const float*)d_in[21];

  // ---- workspace carve-up (~76 MB) ----
  char* wsp = (char*)d_ws;
  auto carve = [&](size_t bytes) {
    char* p = wsp;
    wsp += (bytes + 255) & ~(size_t)255;
    return p;
  };
  __bf16* Wtq  = (__bf16*)carve(256 * 256 * 2);
  __bf16* Wtk  = (__bf16*)carve(256 * 256 * 2);
  __bf16* Wtv  = (__bf16*)carve(256 * 256 * 2);
  __bf16* Wt1  = (__bf16*)carve(256 * 128 * 2);
  __bf16* Wt2  = (__bf16*)carve(256 * 128 * 2);
  __bf16* Wt3  = (__bf16*)carve(256 * 128 * 2);
  __bf16* Wtf  = (__bf16*)carve(256 * 128 * 2);
  __bf16* Wtc  = (__bf16*)carve(128 * 128 * 2);
  float*  Qf32 = (float*) carve((size_t)BN * DKDIM * 4);
  __bf16* Qbf  = (__bf16*)carve((size_t)BN * DKDIM * 2);
  __bf16* Tq   = (__bf16*)carve((size_t)BN * DKDIM * 2);
  __bf16* Kbf  = (__bf16*)carve((size_t)BN * DKDIM * 2);
  __bf16* Tk   = (__bf16*)carve((size_t)BN * DKDIM * 2);
  __bf16* Vt   = (__bf16*)carve((size_t)BN * DKDIM * 2);   // [B][256][NSEQ]
  float*  Qhat = (float*) carve((size_t)BN * DKDIM * 4);

  float* out0 = (float*)d_out;                              // [B][N2][128]
  float* attn = (float*)d_out + (size_t)BN * OUTD;          // [B][N2][N1]

  // ---- 0: weight transposes to bf16 [out][in] ----
  struct { const float* src; __bf16* dst; int fi, fo; } tp[8] = {
    {W_q, Wtq, 256, 256}, {W_k, Wtk, 256, 256}, {W_v, Wtv, 256, 256},
    {W1,  Wt1, 256, 128}, {W2,  Wt2, 256, 128}, {W3,  Wt3, 256, 128},
    {W_f, Wtf, 256, 128}, {W_c, Wtc, 128, 128}};
  for (int t = 0; t < 8; ++t) {
    int total = tp[t].fi * tp[t].fo;
    wt_transpose_kernel<<<(total + 255) / 256, 256, 0, stream>>>(
        tp[t].src, tp[t].dst, tp[t].fi, tp[t].fo);
  }

  // ---- 1: Q / K / V projections (WMMA) ----
  dim3 gblk(BN / 32);
  qkv_gemm_kernel<<<gblk, 256, 0, stream>>>(x2, Wtq, b_q, Qf32, Qbf, Tq, nullptr);
  qkv_gemm_kernel<<<gblk, 256, 0, stream>>>(x1, Wtk, b_k, nullptr, Kbf, Tk, nullptr);
  qkv_gemm_kernel<<<gblk, 256, 0, stream>>>(x1, Wtv, b_v, nullptr, nullptr, nullptr, Vt);

  // ---- 2: attention (scores + softmax + attn output + Q_hat) ----
  const size_t smem = (size_t)16 * SCLD * sizeof(float);    // 256.25 KB of WGP LDS
  hipFuncSetAttribute((const void*)attention_kernel,
                      hipFuncAttributeMaxDynamicSharedMemorySize, (int)smem);
  attention_kernel<<<dim3(NSEQ / 16, BATCH), 256, smem, stream>>>(
      Qbf, Tq, Kbf, Tk, Vt, attn, Qhat);

  // ---- 3: contextual attention gate ----
  cag_kernel<<<dim3(BN / 16), 256, 0, stream>>>(
      Qf32, Qhat, Wt1, b1, Wt2, b2, Wt3, b3, Wtf, b_f, Wtc, b_c,
      g_g, g_b, e_g, e_b, out0);
}
